// NeuralRenderer_39711267618869
// MI455X (gfx1250) — matrix operand order repaired and checked by reading, not compiled, not run on hardware
//
#include <hip/hip_runtime.h>
#include <hip/hip_bf16.h>
#include <math.h>

#define N_RAYS    4096
#define N_SAMPLES 128
#define NPTS      (N_RAYS * N_SAMPLES)
#define NEAR_F    0.5f
#define FAR_F     4.0f
#define BOX_MINF  (-4.0f)
#define BOX_MAXF  (4.0f)
#define HMASK     ((1u << 19) - 1u)
#define CHUNKS    4            // point-groups of 64 per block (amortizes weight staging)

// Column-major f16 weight blob offsets (in halfs): [N][K] layout per matrix
#define OFF_S0 0        // 64 x 32
#define OFF_S1 2048     // 64 x 64
#define OFF_S2 6144     // 16 x 64
#define OFF_C0 7168     // 64 x 64 (rows >=47 zero-padded)
#define OFF_C1 11264    // 64 x 64
#define OFF_C2 15360    // 64 x 64
#define OFF_C3 19456    // 16 x 64 (cols >=3 zero-padded)
#define W_TOTAL 20480

#define RAW_OFFSET (1 << 16)   // float4 raw RGB-sigma buffer lives here in d_ws

typedef __attribute__((ext_vector_type(16))) _Float16 v16h;
typedef __attribute__((ext_vector_type(8)))  float    v8f;

union Frag { v16h h; unsigned u[8]; };

// --- A fragment: 16x32 f16 tile, rows m0..m0+15 of a 32x64 row-major LDS buffer.
// ISA layout: lanes 0-15 -> M=lane, K pairs {0..7,16..23}; lanes 16-31 -> K+8.
__device__ __forceinline__ v16h load_a(const _Float16* buf, int m0, int k0, int lane) {
    const int hs = lane >> 4, m = lane & 15;
    const unsigned* row = (const unsigned*)(buf + (m0 + m) * 64);
    Frag f;
#pragma unroll
    for (int v = 0; v < 8; ++v) {
        const int kb = k0 + ((v < 4) ? (2 * v + 8 * hs) : (16 + 2 * (v - 4) + 8 * hs));
        f.u[v] = row[kb >> 1];
    }
    return f.h;
}

// --- B fragment: 32x16 f16 tile from column-major weights [N][K] with K-stride ks.
// ISA layout: lanes 0-15 -> N=lane, K pairs {0..7,16..23}; lanes 16-31 -> K+8.
__device__ __forceinline__ v16h load_b(const _Float16* w, int ks, int n0, int k0, int lane) {
    const int n = lane & 15, ko = (lane >> 4) * 4;
    const unsigned* col = (const unsigned*)(w + (n0 + n) * ks);
    Frag f;
#pragma unroll
    for (int v = 0; v < 8; ++v) f.u[v] = col[(k0 >> 1) + ((v < 4) ? (v + ko) : (8 + (v - 4) + ko))];
    return f.h;
}

// --- C/D 16x16 f32 store: lane n<16 -> col n rows r; lanes 16-31 -> rows r+8.
__device__ __forceinline__ void store_c(_Float16* dst, int m0, int n0, v8f c,
                                        bool relu, int lane) {
    const int n = lane & 15, mb = (lane >> 4) * 8;
#pragma unroll
    for (int r = 0; r < 8; ++r) {
        float v = c[r];
        if (relu) v = fmaxf(v, 0.0f);
        dst[(m0 + mb + r) * 64 + n0 + n] = (_Float16)v;
    }
}

// GEMM layer: 32 points x K -> 32 points x N. All unique fragments hoisted into
// registers first (one ds_load_b128 pair each), then back-to-back WMMAs.
template <int K, int N, bool RELU>
__device__ __forceinline__ void layer(const _Float16* src, const _Float16* w,
                                      _Float16* dst, int lane) {
    constexpr int KC = K / 32;
    constexpr int NT = N / 16;
    v16h afr[2 * KC];
#pragma unroll
    for (int mt = 0; mt < 2; ++mt)
#pragma unroll
        for (int kc = 0; kc < KC; ++kc)
            afr[mt * KC + kc] = load_a(src, mt * 16, kc * 32, lane);
    v16h bfr[NT * KC];
#pragma unroll
    for (int nt = 0; nt < NT; ++nt)
#pragma unroll
        for (int kc = 0; kc < KC; ++kc)
            bfr[nt * KC + kc] = load_b(w, K, nt * 16, kc * 32, lane);
#pragma unroll
    for (int mt = 0; mt < 2; ++mt)
#pragma unroll
        for (int nt = 0; nt < NT; ++nt) {
            v8f acc = {};
#pragma unroll
            for (int kc = 0; kc < KC; ++kc)
                acc = __builtin_amdgcn_wmma_f32_16x16x32_f16(
                    false, afr[mt * KC + kc], false, bfr[nt * KC + kc],
                    (short)0, acc, false, false);
            store_c(dst, mt * 16, nt * 16, acc, RELU, lane);
        }
}

// ---------------- Kernel 0: fp32 weights -> f16 column-major padded blob ----------
__global__ void nerf_prep_weights(const float* __restrict__ s0, const float* __restrict__ s1,
                                  const float* __restrict__ s2, const float* __restrict__ c0,
                                  const float* __restrict__ c1, const float* __restrict__ c2,
                                  const float* __restrict__ c3, _Float16* __restrict__ blob) {
    const int t = threadIdx.x;
    for (int i = t; i < 64 * 32; i += 256) { int n = i >> 5, k = i & 31; blob[OFF_S0 + i] = (_Float16)s0[k * 64 + n]; }
    for (int i = t; i < 64 * 64; i += 256) { int n = i >> 6, k = i & 63; blob[OFF_S1 + i] = (_Float16)s1[k * 64 + n]; }
    for (int i = t; i < 16 * 64; i += 256) { int n = i >> 6, k = i & 63; blob[OFF_S2 + i] = (_Float16)s2[k * 16 + n]; }
    for (int i = t; i < 64 * 64; i += 256) { int n = i >> 6, k = i & 63; blob[OFF_C0 + i] = (_Float16)((k < 47) ? c0[k * 64 + n] : 0.0f); }
    for (int i = t; i < 64 * 64; i += 256) { int n = i >> 6, k = i & 63; blob[OFF_C1 + i] = (_Float16)c1[k * 64 + n]; }
    for (int i = t; i < 64 * 64; i += 256) { int n = i >> 6, k = i & 63; blob[OFF_C2 + i] = (_Float16)c2[k * 64 + n]; }
    for (int i = t; i < 16 * 64; i += 256) { int n = i >> 6, k = i & 63; blob[OFF_C3 + i] = (_Float16)((n < 3) ? c3[k * 3 + n] : 0.0f); }
}

// ---------------- Kernel 1: per-point featurize + MLPs (WMMA) ---------------------
__global__ __launch_bounds__(64) void nerf_points(
    const float* __restrict__ rays, const float* __restrict__ tables,
    const _Float16* __restrict__ blob, const float* __restrict__ app_table,
    float4* __restrict__ raw) {
    __shared__ _Float16 wlds[W_TOTAL];
    __shared__ _Float16 act[2][2][32 * 64];   // [wave][ping-pong][rows x 64]

    const int tid = threadIdx.x;
    // stage weights into LDS once per block (2560 uint4 across 64 threads)
    {
        const uint4* src = (const uint4*)blob;
        uint4* dst = (uint4*)wlds;
        for (int i = tid; i < (W_TOTAL * 2) / 16; i += 64) dst[i] = src[i];
    }
    __syncthreads();

    const int wv = tid >> 5, lane = tid & 31;
    _Float16* bufA = act[wv][0];
    _Float16* bufB = act[wv][1];

    for (int chunk = 0; chunk < CHUNKS; ++chunk) {
        const int point = (blockIdx.x * CHUNKS + chunk) * 64 + tid;
        const int ray = point >> 7;
        const int smp = point & 127;
        const float ox = rays[ray * 7 + 0], oy = rays[ray * 7 + 1], oz = rays[ray * 7 + 2];
        const float dx = rays[ray * 7 + 3], dy = rays[ray * 7 + 4], dzd = rays[ray * 7 + 5];
        const int code = (int)rays[ray * 7 + 6];

        const float dzstep = (FAR_F - NEAR_F) / (float)(N_SAMPLES - 1);
        const float z = NEAR_F + (float)smp * dzstep;
        float px = ox + dx * z, py = oy + dy * z, pz = oz + dzd * z;
        const bool keep = (px >= BOX_MINF) && (px <= BOX_MAXF) &&
                          (py >= BOX_MINF) && (py <= BOX_MAXF) &&
                          (pz >= BOX_MINF) && (pz <= BOX_MAXF);
        px = fminf(fmaxf(px, BOX_MINF), BOX_MAXF);
        py = fminf(fmaxf(py, BOX_MINF), BOX_MAXF);
        pz = fminf(fmaxf(pz, BOX_MINF), BOX_MAXF);

        // ---- hash-grid encode: 16 levels x 8 corners, L2-resident gathers ----
        {
            const int RES[16] = {16, 20, 25, 32, 40, 50, 64, 80,
                                 101, 128, 161, 203, 256, 322, 406, 512};
#pragma unroll
            for (int l = 0; l < 16; ++l) {
                const float sc = (float)RES[l] * 0.125f;   // res / (BOX_MAX - BOX_MIN)
                const float fx = (px - BOX_MINF) * sc;
                const float fy = (py - BOX_MINF) * sc;
                const float fz = (pz - BOX_MINF) * sc;
                const float bx = floorf(fx), by = floorf(fy), bz = floorf(fz);
                const float rx = fx - bx, ry = fy - by, rz = fz - bz;
                const unsigned ux = (unsigned)bx, uy = (unsigned)by, uz = (unsigned)bz;
                const float* tb = tables + (size_t)l * (size_t)(2u << 19);
                float f0 = 0.0f, f1 = 0.0f;
#pragma unroll
                for (int c = 0; c < 8; ++c) {
                    const unsigned ii = (c >> 2) & 1u, jj = (c >> 1) & 1u, kk = c & 1u;
                    const unsigned h = ((ux + ii) * 1u) ^ ((uy + jj) * 2654435761u) ^
                                       ((uz + kk) * 805459861u);
                    const unsigned idx = h & HMASK;
                    const float wgt = (ii ? rx : 1.0f - rx) * (jj ? ry : 1.0f - ry) *
                                      (kk ? rz : 1.0f - rz);
                    const float2 e = *(const float2*)(tb + (size_t)idx * 2);
                    f0 += wgt * e.x;
                    f1 += wgt * e.y;
                }
                bufA[lane * 64 + 2 * l]     = (_Float16)f0;
                bufA[lane * 64 + 2 * l + 1] = (_Float16)f1;
            }
        }

        // ---- SH-16 view encoding (kept in registers) ----
        float sh[16];
        {
            const float x = dx, y = dy, zz = dzd;
            const float xx = x * x, yy = y * y, z2 = zz * zz;
            const float xy = x * y, yz = y * zz, xz = x * zz;
            sh[0]  = 0.28209479177387814f;
            sh[1]  = -0.4886025119029199f * y;
            sh[2]  = 0.4886025119029199f * zz;
            sh[3]  = -0.4886025119029199f * x;
            sh[4]  = 1.0925484305920792f * xy;
            sh[5]  = -1.0925484305920792f * yz;
            sh[6]  = 0.31539156525252005f * (2.0f * z2 - xx - yy);
            sh[7]  = -1.0925484305920792f * xz;
            sh[8]  = 0.5462742152960396f * (xx - yy);
            sh[9]  = -0.5900435899266435f * y * (3.0f * xx - yy);
            sh[10] = 2.890611442640554f * xy * zz;
            sh[11] = -0.4570457994644658f * y * (4.0f * z2 - xx - yy);
            sh[12] = 0.3731763325901154f * zz * (2.0f * z2 - 3.0f * xx - 3.0f * yy);
            sh[13] = -0.4570457994644658f * x * (4.0f * z2 - xx - yy);
            sh[14] = 1.445305721320277f * zz * (xx - yy);
            sh[15] = -0.5900435899266435f * x * (xx - 3.0f * yy);
        }

        // ---- sigma MLP: 32 -> 64 -> 64 -> 16 (WMMA f32_16x16x32_f16) ----
        layer<32, 64, true >(bufA, wlds + OFF_S0, bufB, lane);
        layer<64, 64, true >(bufB, wlds + OFF_S1, bufA, lane);
        layer<64, 16, false>(bufA, wlds + OFF_S2, bufB, lane);
        const float sigma = keep ? (float)bufB[lane * 64 + 0] : 0.0f;

        // ---- assemble color input: [SH(16) | geo(15) | app(16) | pad(17)] ----
#pragma unroll
        for (int j = 0; j < 16; ++j) bufA[lane * 64 + j] = (_Float16)sh[j];
#pragma unroll
        for (int j = 0; j < 15; ++j) bufA[lane * 64 + 16 + j] = bufB[lane * 64 + 1 + j];
        const float* ap = app_table + code * 16;
#pragma unroll
        for (int j = 0; j < 16; ++j) bufA[lane * 64 + 31 + j] = (_Float16)ap[j];
#pragma unroll
        for (int j = 47; j < 64; ++j) bufA[lane * 64 + j] = (_Float16)0.0f;

        // ---- color MLP: 64 -> 64 -> 64 -> 64 -> 3 ----
        layer<64, 64, true >(bufA, wlds + OFF_C0, bufB, lane);
        layer<64, 64, true >(bufB, wlds + OFF_C1, bufA, lane);
        layer<64, 64, true >(bufA, wlds + OFF_C2, bufB, lane);
        layer<64, 16, false>(bufB, wlds + OFF_C3, bufA, lane);

        raw[point] = make_float4((float)bufA[lane * 64 + 0],
                                 (float)bufA[lane * 64 + 1],
                                 (float)bufA[lane * 64 + 2], sigma);
    }
}

// ---------------- Kernel 2: per-ray alpha compositing -----------------------------
__global__ __launch_bounds__(128) void nerf_composite(
    const float* __restrict__ rays, const float4* __restrict__ raw,
    float* __restrict__ out) {
    const int r = blockIdx.x * 128 + threadIdx.x;
    if (r >= N_RAYS) return;
    const float dx = rays[r * 7 + 3], dy = rays[r * 7 + 4], dz = rays[r * 7 + 5];
    const float nrm = sqrtf(dx * dx + dy * dy + dz * dz);
    const float dzstep = (FAR_F - NEAR_F) / (float)(N_SAMPLES - 1);

    float T = 1.0f;
    float c0 = 0.0f, c1 = 0.0f, c2 = 0.0f, depth = 0.0f;
    float sumW = 0.0f, Q = 0.0f, E = 0.0f;
#pragma unroll 4
    for (int s = 0; s < N_SAMPLES; ++s) {
        const float4 rv = raw[r * N_SAMPLES + s];
        const float dist = ((s == N_SAMPLES - 1) ? 1e10f : dzstep) * nrm;
        const float sg = fmaxf(rv.w, 0.0f);
        const float alpha = 1.0f - expf(-sg * dist);
        const float w = alpha * T;
        T *= (1.0f - alpha + 1e-10f);
        const float z = NEAR_F + (float)s * dzstep;
        c0 += w * (1.0f / (1.0f + expf(-rv.x)));
        c1 += w * (1.0f / (1.0f + expf(-rv.y)));
        c2 += w * (1.0f / (1.0f + expf(-rv.z)));
        depth += w * z;
        sumW += w;
        const float q = fmaxf(w, 1e-12f);
        Q += q;
        E += q * logf(q);
    }
    const float ql = fmaxf(1.0f - sumW + 1e-6f, 1e-12f);
    Q += ql;
    E += ql * logf(ql);

    out[r * 3 + 0] = c0;
    out[r * 3 + 1] = c1;
    out[r * 3 + 2] = c2;
    out[N_RAYS * 3 + r] = depth;                 // depth_map
    out[N_RAYS * 4 + r] = logf(Q) - E / Q;       // sparsity (entropy)
}

extern "C" void kernel_launch(void* const* d_in, const int* in_sizes, int n_in,
                              void* d_out, int out_size, void* d_ws, size_t ws_size,
                              hipStream_t stream) {
    (void)in_sizes; (void)n_in; (void)out_size; (void)ws_size;
    const float* rays      = (const float*)d_in[0];
    // d_in[1] = app_code (unused by the reference forward)
    const float* tables    = (const float*)d_in[2];
    const float* w_s0      = (const float*)d_in[3];
    const float* w_s1      = (const float*)d_in[4];
    const float* w_s2      = (const float*)d_in[5];
    const float* w_c0      = (const float*)d_in[6];
    const float* w_c1      = (const float*)d_in[7];
    const float* w_c2      = (const float*)d_in[8];
    const float* w_c3      = (const float*)d_in[9];
    const float* app_table = (const float*)d_in[10];

    _Float16* blob = (_Float16*)d_ws;
    float4* raw = (float4*)((char*)d_ws + RAW_OFFSET);

    nerf_prep_weights<<<1, 256, 0, stream>>>(w_s0, w_s1, w_s2, w_c0, w_c1, w_c2, w_c3, blob);
    nerf_points<<<NPTS / (64 * CHUNKS), 64, 0, stream>>>(rays, tables, blob, app_table, raw);
    nerf_composite<<<N_RAYS / 128, 128, 0, stream>>>(rays, raw, (float*)d_out);
}
